// ImprovedSpeMamba_NoMultiScaleConv_20298015441626
// MI455X (gfx1250) — matrix operand order repaired and checked by reading, not compiled, not run on hardware
//
#include <hip/hip_runtime.h>
#include <hip/hip_bf16.h>

typedef __attribute__((ext_vector_type(16))) __bf16 v16bf;
typedef __attribute__((ext_vector_type(8)))  __bf16 v8bf;
typedef __attribute__((ext_vector_type(2)))  __bf16 v2bf;
typedef __attribute__((ext_vector_type(8)))  float  v8f;

// Problem constants
#define NB   2
#define NC   128
#define NH   128
#define NW   128
#define TOT  (NB*NC*NH*NW)        // 4,194,304
#define GRPN (TOT/8)              // elems per (b,group) = 524288

#if __has_builtin(__builtin_amdgcn_sched_barrier)
#define SCHED_FENCE() __builtin_amdgcn_sched_barrier(0)
#else
#define SCHED_FENCE()
#endif

__device__ __forceinline__ float silu_f(float v) {
    return v / (1.f + __expf(-v));
}
__device__ __forceinline__ float softplus_f(float v) {
    if (v > 20.f)  return v;
    if (v < -20.f) return __expf(v);
    return __logf(1.f + __expf(v));
}

// ---------------------------------------------------------------------------
// Kernel 0: pre-pack conv weights into WMMA-B fragment order (bf16).
// B frag for (tap, ciChunk32, nTile): lane holds N = nTile*16 + lane%16,
// K = (lane<16 ? {0..7,16..23} : {8..15,24..31}), 2 per VGPR -> 32 contiguous bytes per lane.
// flat: ((((tap*4+chunk)*8+nt)*32+lane)*16+j)
// ---------------------------------------------------------------------------
__global__ __launch_bounds__(256) void prep_weights(
    const float* __restrict__ conv_w, __bf16* __restrict__ WB) {
    int idx = blockIdx.x * 256 + threadIdx.x;
    if (idx >= 9 * 4 * 8 * 32 * 16) return;
    int j     = idx & 15;
    int lane  = (idx >> 4) & 31;
    int nt    = (idx >> 9) & 7;
    int chunk = (idx >> 12) & 3;
    int tap   = idx >> 14;
    int co = nt * 16 + (lane & 15);
    int ci = chunk * 32 + ((lane & 16) ? 8 : 0) + (j & 7) + ((j & 8) ? 16 : 0);
    WB[idx] = (__bf16)conv_w[co * 1152 + ci * 9 + tap];   // (co,ci,kh,kw)
}

// ---------------------------------------------------------------------------
// Kernel 1: 3x3 conv as implicit GEMM with v_wmma_f32_16x16x32_bf16.
// One block per (h,b): 128(w) x 128(co) output tile, K = 9 taps x 128 ci.
// 8 waves, wave r owns w-rows [16r,16r+16), 8 accumulators (full co=128).
// A sched_barrier between the fragment loads and the WMMAs forces all 8 B
// fragments live at once -> distinct registers, partial loadcnt waits, and
// back-to-back WMMA issue.
// ---------------------------------------------------------------------------
__global__ __launch_bounds__(256) void conv3x3_wmma(
    const float* __restrict__ x, const float* __restrict__ conv_b,
    const __bf16* __restrict__ WB, float* __restrict__ y) {
    const int h    = blockIdx.x;
    const int b    = blockIdx.y;
    const int tid  = threadIdx.x;
    const int lane = tid & 31;
    const int wave = tid >> 5;
    const int mBase = wave * 16;

    // Xs[dh=3][wi=130][ci_local 64 (+pad to 72 for 16B row alignment)]
    __shared__ __align__(16) __bf16 Xs[3 * 130 * 72];

    const v8f vzero = {0.f, 0.f, 0.f, 0.f, 0.f, 0.f, 0.f, 0.f};
    v8f acc[8];
#pragma unroll
    for (int i = 0; i < 8; ++i) acc[i] = vzero;

    const v16bf* bp = (const v16bf*)WB;

    for (int half = 0; half < 2; ++half) {
        __syncthreads();
        // stage 3 input rows x 64 ci x 130 w (with halo), f32 -> packed bf16x2
        for (int idx = tid; idx < 3 * 32 * 130; idx += 256) {
            int dh = idx / (32 * 130);
            int r  = idx % (32 * 130);
            int cp = r / 130;                   // ci pair
            int wi = r % 130;
            int hs  = h + dh - 1;
            int wsx = wi - 1;
            float v0 = 0.f, v1 = 0.f;
            if ((unsigned)hs < 128u && (unsigned)wsx < 128u) {
                const float* src =
                    &x[((b * 128 + half * 64 + cp * 2) * 128 + hs) * 128 + wsx];
                v0 = src[0];
                v1 = src[128 * 128];            // next ci plane
            }
            v2bf p;
            p[0] = (__bf16)v0;
            p[1] = (__bf16)v1;
            *(v2bf*)&Xs[(dh * 130 + wi) * 72 + cp * 2] = p;
        }
        __syncthreads();

        for (int tap = 0; tap < 9; ++tap) {
            const int kh = tap / 3, kw = tap % 3;
#pragma unroll
            for (int ck = 0; ck < 2; ++ck) {
                // A fragment: lanes 0..15 -> K {0..7,16..23}, 16..31 -> complement
                const int wi = mBase + (lane & 15) + kw;
                const int e0 = ck * 32 + ((lane & 16) ? 8 : 0);
                const __bf16* ap = &Xs[(kh * 130 + wi) * 72 + e0];
                v8bf alo = *(const v8bf*)ap;
                v8bf ahi = *(const v8bf*)(ap + 16);
                v16bf a;
#pragma unroll
                for (int i = 0; i < 8; ++i) { a[i] = alo[i]; a[i + 8] = ahi[i]; }

                const int gchunk = half * 2 + ck;
                const int bBase  = (tap * 4 + gchunk) * 8;

                // preload ALL 8 B fragments into distinct registers
                v16bf bfr[8];
#pragma unroll
                for (int nt = 0; nt < 8; ++nt)
                    bfr[nt] = bp[(bBase + nt) * 32 + lane];

                // fence: no load may sink past here, no WMMA may hoist above
                SCHED_FENCE();

                // 8 back-to-back WMMAs (shared A, distinct B/D: no RAW hazard)
#pragma unroll
                for (int nt = 0; nt < 8; ++nt)
                    acc[nt] = __builtin_amdgcn_wmma_f32_16x16x32_bf16(
                        false, a, false, bfr[nt], (short)0, acc[nt], false, false);

                SCHED_FENCE();
            }
        }
    }

    // epilogue: C/D layout -> lane<16: M=v,N=lane ; lane>=16: M=v+8,N=lane-16
    const int wrow = mBase + ((lane & 16) ? 8 : 0);
#pragma unroll
    for (int nt = 0; nt < 8; ++nt) {
        const int co = nt * 16 + (lane & 15);
        const float bias = conv_b[co];
        const int base = ((b * 128 + co) * 128 + h) * 128;
#pragma unroll
        for (int v = 0; v < 8; ++v)
            y[base + wrow + v] = acc[nt][v] + bias;
    }
}

// ---------------------------------------------------------------------------
// Kernel 2: Mamba over each image row. One wave32 per sequence (lane = d_inner
// channel, D_INNER==32), 8 sequences per 256-thread block. In-place on y.
// ---------------------------------------------------------------------------
__global__ __launch_bounds__(256) void mamba_kernel(
    float* __restrict__ y,
    const float* __restrict__ in_proj_w,   // (64,16)
    const float* __restrict__ conv1d_w,    // (32,1,4)
    const float* __restrict__ conv1d_b,    // (32)
    const float* __restrict__ x_proj_w,    // (33,32)
    const float* __restrict__ dt_proj_w,   // (32,1)
    const float* __restrict__ dt_proj_b,   // (32)
    const float* __restrict__ A_log,       // (32,16)
    const float* __restrict__ Dp,          // (32)
    const float* __restrict__ out_proj_w)  // (16,32)
{
    const int tid  = threadIdx.x;
    const int lane = tid & 31;
    const int wv   = tid >> 5;
    const long n    = (long)blockIdx.x * 8 + wv;
    const long base = n * 128;

    __shared__ float seqs[8][128];
    __shared__ float xcs[8][8][32];    // also reused for ym
    __shared__ float xdbls[8][8][33];

    *(float4*)&seqs[wv][lane * 4] = *(const float4*)(y + base + lane * 4);
    __syncthreads();

    const int d = lane;
    // ---- in_proj ----
    float wxi[16], wz[16];
#pragma unroll
    for (int g = 0; g < 16; ++g) {
        wxi[g] = in_proj_w[d * 16 + g];
        wz[g]  = in_proj_w[(32 + d) * 16 + g];
    }
    float xi[8], z[8];
#pragma unroll
    for (int t = 0; t < 8; ++t) {
        float sx = 0.f, sz = 0.f;
#pragma unroll
        for (int g = 0; g < 16; ++g) {
            float s = seqs[wv][t * 16 + g];
            sx += s * wxi[g];
            sz += s * wz[g];
        }
        xi[t] = sx; z[t] = sz;
    }
    // ---- causal depthwise conv1d (k=4) + SiLU ----
    float c1[4];
#pragma unroll
    for (int k = 0; k < 4; ++k) c1[k] = conv1d_w[d * 4 + k];
    const float cb = conv1d_b[d];
    float xc[8];
#pragma unroll
    for (int t = 0; t < 8; ++t) {
        float s = cb;
#pragma unroll
        for (int k = 0; k < 4; ++k) {
            int tt = t - 3 + k;
            if (tt >= 0) s += c1[k] * xi[tt];
        }
        xc[t] = silu_f(s);
    }
#pragma unroll
    for (int t = 0; t < 8; ++t) xcs[wv][t][d] = xc[t];
    __syncthreads();

    // ---- x_proj: 33 outputs per token, lane j computes column j (+32 on lane 0)
#pragma unroll
    for (int rep = 0; rep < 2; ++rep) {
        int j = lane + rep * 32;
        if (j < 33) {
#pragma unroll
            for (int t = 0; t < 8; ++t) {
                float s = 0.f;
#pragma unroll
                for (int dd = 0; dd < 32; ++dd)
                    s += xcs[wv][t][dd] * x_proj_w[j * 32 + dd];
                xdbls[wv][t][j] = s;
            }
        }
    }
    __syncthreads();

    // ---- selective scan: h[16] per lane ----
    float aS[16], hS[16];
#pragma unroll
    for (int s = 0; s < 16; ++s) { aS[s] = -__expf(A_log[d * 16 + s]); hS[s] = 0.f; }
    const float dtw = dt_proj_w[d], dtb = dt_proj_b[d], Dd = Dp[d];
    float ym[8];
#pragma unroll
    for (int t = 0; t < 8; ++t) {
        float dt = softplus_f(dtw * xdbls[wv][t][0] + dtb);
        float dx = dt * xc[t];
        float yt = 0.f;
#pragma unroll
        for (int s = 0; s < 16; ++s) {
            float dA = __expf(dt * aS[s]);
            hS[s] = dA * hS[s] + dx * xdbls[wv][t][1 + s];
            yt += hS[s] * xdbls[wv][t][17 + s];
        }
        ym[t] = (yt + Dd * xc[t]) * silu_f(z[t]);
    }
    __syncthreads();
#pragma unroll
    for (int t = 0; t < 8; ++t) xcs[wv][t][d] = ym[t];   // reuse as ym buffer
    __syncthreads();

    // ---- out_proj: 128 outputs per sequence, 4 per lane ----
#pragma unroll
    for (int i = 0; i < 4; ++i) {
        int idx = lane + 32 * i;
        int t = idx >> 4, g = idx & 15;
        float s = 0.f;
#pragma unroll
        for (int dd = 0; dd < 32; ++dd)
            s += xcs[wv][t][dd] * out_proj_w[g * 32 + dd];
        y[base + idx] = s;
    }
}

// ---------------------------------------------------------------------------
// Kernel 3a/3b: deterministic GroupNorm stats (block partials + tree reduce).
// Group (b,g) is a contiguous 524288-element span -> grp = flat / 524288.
// ---------------------------------------------------------------------------
__global__ __launch_bounds__(256) void gn_partials(
    const float* __restrict__ ym, float* __restrict__ partial) {
    __shared__ float s1[256], s2[256];
    const int tid = threadIdx.x;
    const long base = (long)blockIdx.x * 4096;
    float a = 0.f, b2 = 0.f;
#pragma unroll
    for (int k = 0; k < 16; ++k) {
        float v = ym[base + tid + k * 256];
        a += v; b2 += v * v;
    }
    s1[tid] = a; s2[tid] = b2;
    __syncthreads();
    for (int off = 128; off > 0; off >>= 1) {
        if (tid < off) { s1[tid] += s1[tid + off]; s2[tid] += s2[tid + off]; }
        __syncthreads();
    }
    if (tid == 0) {
        partial[blockIdx.x * 2]     = s1[0];
        partial[blockIdx.x * 2 + 1] = s2[0];
    }
}

__global__ __launch_bounds__(128) void gn_reduce(
    const float* __restrict__ partial, float* __restrict__ stats) {
    __shared__ float s1[128], s2[128];
    const int tid = threadIdx.x;
    const int grp = blockIdx.x;               // 8 groups, 128 partials each
    s1[tid] = partial[(grp * 128 + tid) * 2];
    s2[tid] = partial[(grp * 128 + tid) * 2 + 1];
    __syncthreads();
    for (int off = 64; off > 0; off >>= 1) {
        if (tid < off) { s1[tid] += s1[tid + off]; s2[tid] += s2[tid + off]; }
        __syncthreads();
    }
    if (tid == 0) {
        stats[grp * 2]     = s1[0];
        stats[grp * 2 + 1] = s2[0];
    }
}

__global__ __launch_bounds__(256) void gn_finalize(
    const float* __restrict__ ym, const float* __restrict__ x,
    const float* __restrict__ gn_w, const float* __restrict__ gn_b,
    const float* __restrict__ stats, float* __restrict__ out) {
    const long e = ((long)blockIdx.x * 256 + threadIdx.x) * 4;
    const int c   = (int)((e >> 14) & 127);
    const int grp = (int)(e / GRPN);
    const float mean = stats[grp * 2] * (1.f / (float)GRPN);
    const float var  = stats[grp * 2 + 1] * (1.f / (float)GRPN) - mean * mean;
    const float rstd = rsqrtf(var + 1e-5f);
    const float gw = gn_w[c] * rstd, gb = gn_b[c];
    float4 v  = *(const float4*)(ym + e);
    float4 xv = *(const float4*)(x + e);
    float vv[4] = {v.x, v.y, v.z, v.w};
    float xx[4] = {xv.x, xv.y, xv.z, xv.w};
    float oo[4];
#pragma unroll
    for (int k = 0; k < 4; ++k) {
        float xp = (vv[k] - mean) * gw + gb;
        oo[k] = xx[k] + xp / (1.f + __expf(-xp));
    }
    float4 o4 = {oo[0], oo[1], oo[2], oo[3]};
    *(float4*)(out + e) = o4;
}

// ---------------------------------------------------------------------------
extern "C" void kernel_launch(void* const* d_in, const int* in_sizes, int n_in,
                              void* d_out, int out_size, void* d_ws, size_t ws_size,
                              hipStream_t stream) {
    (void)in_sizes; (void)n_in; (void)out_size; (void)ws_size;
    const float* x       = (const float*)d_in[0];
    const float* conv_w  = (const float*)d_in[1];
    const float* conv_b  = (const float*)d_in[2];
    const float* gn_w    = (const float*)d_in[3];
    const float* gn_b    = (const float*)d_in[4];
    const float* in_proj = (const float*)d_in[5];
    const float* c1w     = (const float*)d_in[6];
    const float* c1b     = (const float*)d_in[7];
    const float* xpw     = (const float*)d_in[8];
    const float* dtw     = (const float*)d_in[9];
    const float* dtb     = (const float*)d_in[10];
    const float* alog    = (const float*)d_in[11];
    const float* Dp      = (const float*)d_in[12];
    const float* outw    = (const float*)d_in[13];

    // workspace layout
    char* ws = (char*)d_ws;
    float*  y       = (float*)ws;                              // 16 MB (y, then mamba out in-place)
    __bf16* WB      = (__bf16*)(ws + (size_t)TOT * 4);         // 294,912 B packed weights
    float*  partial = (float*)(ws + (size_t)TOT * 4 + 294912); // 1024*2 f32
    float*  stats   = partial + 2048;                          // 8*2 f32

    prep_weights<<<576, 256, 0, stream>>>(conv_w, WB);
    conv3x3_wmma<<<dim3(NH, NB), 256, 0, stream>>>(x, conv_b, WB, y);
    mamba_kernel<<<4096, 256, 0, stream>>>(y, in_proj, c1w, c1b, xpw,
                                           dtw, dtb, alog, Dp, outw);
    gn_partials<<<1024, 256, 0, stream>>>(y, partial);
    gn_reduce<<<8, 128, 0, stream>>>(partial, stats);
    gn_finalize<<<4096, 256, 0, stream>>>(y, x, gn_w, gn_b, stats, (float*)d_out);
}
